// Kalman_Filter_29162827940570
// MI455X (gfx1250) — compile-verified, matrix operand order checked by reading, not run on hardware
//
#include <hip/hip_runtime.h>

typedef __attribute__((ext_vector_type(16))) _Float16 v16h;
typedef __attribute__((ext_vector_type(8)))  float    v8f;
typedef __attribute__((ext_vector_type(2)))  float    v2f;

#define DZv  32
#define DAv  16
#define KMIX 3
#define TT   128
#define BSv  512
#define HH   128
#define G4   512   // 4*H

// ---------------- WMMA fragment index helpers (CDNA5 ISA 7.12.2) --
// 16-bit A-matrix 16x32 (MxK): lane = 16*half + m ; element j -> K index.
// Per lane this is two contiguous 8-half runs -> compiler emits 2x ds_load_b128.
__device__ __host__ __forceinline__ int kmap16(int lane, int j) {
  int half = (lane >> 4) & 1;
  int v = j >> 1, pos = j & 1;
  return (v < 4) ? (2 * v + pos + 8 * half) : (16 + 2 * (v - 4) + pos + 8 * half);
}

// -------- packed weight fragment table (B = W^T tiles, 32x16 per frag) ------
#define FR_L0IH 0
#define NF_L0IH 32            // 32 n-tiles, 1 k-tile (K=32, padded from 16)
#define FR_L0HH (FR_L0IH + NF_L0IH)   // 32
#define NF_L0HH (32 * 4)
#define FR_L1IH (FR_L0HH + NF_L0HH)   // 160
#define NF_L1IH (32 * 4)
#define FR_L1HH (FR_L1IH + NF_L1IH)   // 288
#define NF_L1HH (32 * 4)
#define NFRAG   (FR_L1HH + NF_L1HH)   // 416 fragments * 32 lanes * 16 halves

__global__ void pack_weights(const float* __restrict__ Wih0,
                             const float* __restrict__ Whh0,
                             const float* __restrict__ Wih1,
                             const float* __restrict__ Whh1,
                             _Float16* __restrict__ packed) {
  int gid = blockIdx.x * blockDim.x + threadIdx.x;
  if (gid >= NFRAG * 32) return;
  int f = gid >> 5;
  int lane = gid & 31;
  const float* W; int din; int nt, kt;
  if (f < FR_L0HH)      { W = Wih0; din = 16;  nt = f; kt = 0; }
  else if (f < FR_L1IH) { W = Whh0; din = 128; int i = f - FR_L0HH; nt = i >> 2; kt = i & 3; }
  else if (f < FR_L1HH) { W = Wih1; din = 128; int i = f - FR_L1IH; nt = i >> 2; kt = i & 3; }
  else                  { W = Whh1; din = 128; int i = f - FR_L1HH; nt = i >> 2; kt = i & 3; }
  int n = nt * 16 + (lane & 15);
  _Float16* dst = packed + ((size_t)f * 32 + lane) * 16;
#pragma unroll
  for (int j = 0; j < 16; ++j) {
    int k = kt * 32 + kmap16(lane, j);
    float v = (k < din) ? W[n * din + k] : 0.0f;
    dst[j] = (_Float16)v;
  }
}

__global__ void pack_bias(const float* bih0, const float* bhh0,
                          const float* bih1, const float* bhh1,
                          float* bias0, float* bias1) {
  int i = blockIdx.x * blockDim.x + threadIdx.x;
  if (i < G4) { bias0[i] = bih0[i] + bhh0[i]; bias1[i] = bih1[i] + bhh1[i]; }
}

__device__ __forceinline__ v16h ldfrag(const _Float16* __restrict__ packed, int f, int lane) {
  return *(const v16h*)(packed + ((size_t)f * 32 + lane) * 16);
}

// ---------------- phase 1: 2-layer LSTM + mixture weights alpha -------------
// Block: 1024 threads = 32 waves = (ks in [0,4)) x (g in [0,8)), 16 batch rows.
// Wave (ks,g) computes gate columns [g*16,(g+1)*16) of all four gates, K-slice
// ks only -> at most 16 resident weight fragments per wave (128 VGPRs), so the
// weights can stay in registers across all T steps. Partial gate sums are
// reduced through LDS (ks==0 stores, ks!=0 ds_add_f32). ks==0 waves run the
// cell update on co-located i/f/g/o tiles; c stays in registers; h -> LDS f16.
__global__ void __launch_bounds__(1024)
lstm_alpha(const float* __restrict__ a, const float* __restrict__ a0,
           const _Float16* __restrict__ packed,
           const float* __restrict__ bias0, const float* __restrict__ bias1,
           const float* __restrict__ Wlin, const float* __restrict__ blin,
           float* __restrict__ alpha_out) {
  __shared__ _Float16 xs[16][32];     // x_t, K-padded to 32
  __shared__ _Float16 h0s[16][HH];
  __shared__ _Float16 h1s[16][HH];
  __shared__ float    gbuf[4][16][HH];  // gate pre-activations [gate][row][col]
  __shared__ float    lgbuf[16][KMIX];

  const int tid   = threadIdx.x;
  const int lane  = tid & 31;
  const int wave  = tid >> 5;         // 0..31
  const int ks    = wave >> 3;        // K-slice 0..3
  const int g     = wave & 7;         // gate-column group
  const int n16   = lane & 15;
  const int half8 = (lane >> 4) * 8;
  const int gcol  = g * 16 + n16;
  const int b0    = blockIdx.x * 16;

  for (int i = tid; i < 16 * 32; i += 1024) ((_Float16*)xs)[i] = (_Float16)0.0f;
  for (int i = tid; i < 16 * HH; i += 1024) {
    ((_Float16*)h0s)[i] = (_Float16)0.0f;
    ((_Float16*)h1s)[i] = (_Float16)0.0f;
  }
  float c0[8], c1[8];
#pragma unroll
  for (int e = 0; e < 8; ++e) { c0[e] = 0.0f; c1[e] = 0.0f; }

  // per-lane gate biases (used by the ks==0 update waves)
  float bia0 = bias0[0 * HH + gcol], bfa0 = bias0[1 * HH + gcol];
  float bga0 = bias0[2 * HH + gcol], boa0 = bias0[3 * HH + gcol];
  float bia1 = bias1[0 * HH + gcol], bfa1 = bias1[1 * HH + gcol];
  float bga1 = bias1[2 * HH + gcol], boa1 = bias1[3 * HH + gcol];

  // resident weight fragments (<=16 per wave; loaded once)
  v16h wl0a[4], wl0b[4], wl1a[4], wl1b[4];
#pragma unroll
  for (int q = 0; q < 4; ++q) {
    if (ks == 0) {
      wl0a[q] = ldfrag(packed, FR_L0IH + q * 8 + g, lane);
      wl0b[q] = ldfrag(packed, FR_L0HH + (q * 8 + g) * 4 + 0, lane);
    } else {
      wl0a[q] = ldfrag(packed, FR_L0HH + (q * 8 + g) * 4 + ks, lane);
      wl0b[q] = wl0a[q];  // unused on this path; keep defined
    }
    wl1a[q] = ldfrag(packed, FR_L1IH + (q * 8 + g) * 4 + ks, lane);
    wl1b[q] = ldfrag(packed, FR_L1HH + (q * 8 + g) * 4 + ks, lane);
  }
  __syncthreads();

  for (int t = 0; t < TT; ++t) {
    // x_t = (t==0 ? a0 : a[:, t-1]) into cols [0,16); cols [16,32) stay zero
    if (tid < 256) {
      int b = tid >> 4, d = tid & 15;
      float v = (t == 0) ? a0[d] : a[((size_t)(b0 + b) * TT + (t - 1)) * DAv + d];
      xs[b][d] = (_Float16)v;
    }
    __syncthreads();

    // ---- layer 0 partial gates (K-slice ks) ----
    v8f acc[4];
#pragma unroll
    for (int q = 0; q < 4; ++q) acc[q] = (v8f){};
    if (ks == 0) {
      v16h ax, ah;
#pragma unroll
      for (int j = 0; j < 16; ++j) ax[j] = xs[n16][kmap16(lane, j)];
#pragma unroll
      for (int j = 0; j < 16; ++j) ah[j] = h0s[n16][kmap16(lane, j)];
#pragma unroll
      for (int q = 0; q < 4; ++q)
        acc[q] = __builtin_amdgcn_wmma_f32_16x16x32_f16(false, ax, false, wl0a[q],
                                                        (short)0, acc[q], false, false);
#pragma unroll
      for (int q = 0; q < 4; ++q)
        acc[q] = __builtin_amdgcn_wmma_f32_16x16x32_f16(false, ah, false, wl0b[q],
                                                        (short)0, acc[q], false, false);
    } else {
      v16h ah;
#pragma unroll
      for (int j = 0; j < 16; ++j) ah[j] = h0s[n16][ks * 32 + kmap16(lane, j)];
#pragma unroll
      for (int q = 0; q < 4; ++q)
        acc[q] = __builtin_amdgcn_wmma_f32_16x16x32_f16(false, ah, false, wl0a[q],
                                                        (short)0, acc[q], false, false);
    }
    // reduce partials: ks==0 initializes, others accumulate atomically
    if (ks == 0) {
#pragma unroll
      for (int q = 0; q < 4; ++q)
#pragma unroll
        for (int r = 0; r < 8; ++r) gbuf[q][half8 + r][gcol] = acc[q][r];
    }
    __syncthreads();
    if (ks != 0) {
#pragma unroll
      for (int q = 0; q < 4; ++q)
#pragma unroll
        for (int r = 0; r < 8; ++r) atomicAdd(&gbuf[q][half8 + r][gcol], acc[q][r]);
    }
    __syncthreads();
    // cell update on ks==0 waves (gate order i,f,g,o)
    if (ks == 0) {
#pragma unroll
      for (int r = 0; r < 8; ++r) {
        float gi = gbuf[0][half8 + r][gcol] + bia0;
        float gf = gbuf[1][half8 + r][gcol] + bfa0;
        float gc = gbuf[2][half8 + r][gcol] + bga0;
        float go = gbuf[3][half8 + r][gcol] + boa0;
        float si = 1.0f / (1.0f + __expf(-gi));
        float sf = 1.0f / (1.0f + __expf(-gf));
        float so = 1.0f / (1.0f + __expf(-go));
        c0[r] = sf * c0[r] + si * tanhf(gc);
        h0s[half8 + r][gcol] = (_Float16)(so * tanhf(c0[r]));
      }
    }
    __syncthreads();

    // ---- layer 1 partial gates (K-slice ks) ----
#pragma unroll
    for (int q = 0; q < 4; ++q) acc[q] = (v8f){};
    {
      v16h ah0, ah1;
#pragma unroll
      for (int j = 0; j < 16; ++j) ah0[j] = h0s[n16][ks * 32 + kmap16(lane, j)];
#pragma unroll
      for (int j = 0; j < 16; ++j) ah1[j] = h1s[n16][ks * 32 + kmap16(lane, j)];
#pragma unroll
      for (int q = 0; q < 4; ++q)
        acc[q] = __builtin_amdgcn_wmma_f32_16x16x32_f16(false, ah0, false, wl1a[q],
                                                        (short)0, acc[q], false, false);
#pragma unroll
      for (int q = 0; q < 4; ++q)
        acc[q] = __builtin_amdgcn_wmma_f32_16x16x32_f16(false, ah1, false, wl1b[q],
                                                        (short)0, acc[q], false, false);
    }
    if (ks == 0) {
#pragma unroll
      for (int q = 0; q < 4; ++q)
#pragma unroll
        for (int r = 0; r < 8; ++r) gbuf[q][half8 + r][gcol] = acc[q][r];
    }
    __syncthreads();
    if (ks != 0) {
#pragma unroll
      for (int q = 0; q < 4; ++q)
#pragma unroll
        for (int r = 0; r < 8; ++r) atomicAdd(&gbuf[q][half8 + r][gcol], acc[q][r]);
    }
    __syncthreads();
    if (ks == 0) {
#pragma unroll
      for (int r = 0; r < 8; ++r) {
        float gi = gbuf[0][half8 + r][gcol] + bia1;
        float gf = gbuf[1][half8 + r][gcol] + bfa1;
        float gc = gbuf[2][half8 + r][gcol] + bga1;
        float go = gbuf[3][half8 + r][gcol] + boa1;
        float si = 1.0f / (1.0f + __expf(-gi));
        float sf = 1.0f / (1.0f + __expf(-gf));
        float so = 1.0f / (1.0f + __expf(-go));
        c1[r] = sf * c1[r] + si * tanhf(gc);
        h1s[half8 + r][gcol] = (_Float16)(so * tanhf(c1[r]));
      }
    }
    __syncthreads();

    // alpha = softmax(h1 @ Wlin^T + blin)
    if (tid < 16 * KMIX) {
      int b = tid / KMIX, k = tid % KMIX;
      float s = blin[k];
      for (int j = 0; j < HH; ++j) s += (float)h1s[b][j] * Wlin[k * HH + j];
      lgbuf[b][k] = s;
    }
    __syncthreads();
    if (tid < 16) {
      float l0 = lgbuf[tid][0], l1 = lgbuf[tid][1], l2 = lgbuf[tid][2];
      float m  = fmaxf(l0, fmaxf(l1, l2));
      float e0 = __expf(l0 - m), e1 = __expf(l1 - m), e2 = __expf(l2 - m);
      float inv = 1.0f / (e0 + e1 + e2);
      float* dst = alpha_out + ((size_t)(b0 + tid) * TT + t) * KMIX;
      dst[0] = e0 * inv; dst[1] = e1 * inv; dst[2] = e2 * inv;
    }
    __syncthreads();
  }
}

// ---------------- f32 WMMA 16x16x4 tile helpers for the Kalman phase --------
__device__ __forceinline__ v2f ldA4(const float* M, int ld, int m0, int k0, int lane) {
  int m = m0 + (lane & 15);
  int k = k0 + ((lane >> 4) << 1);
  v2f a; a.x = M[m * ld + k]; a.y = M[m * ld + k + 1];
  return a;
}
__device__ __forceinline__ v2f ldB4(const float* M, int ld, int k0, int n0, int lane) {
  int n = n0 + (lane & 15);
  int k = k0 + ((lane >> 4) << 1);
  v2f b; b.x = M[k * ld + n]; b.y = M[(k + 1) * ld + n];
  return b;
}
__device__ __forceinline__ v2f ldBT4(const float* M, int ld, int k0, int n0, int lane) {
  int n = n0 + (lane & 15);
  int k = k0 + ((lane >> 4) << 1);
  v2f b; b.x = M[n * ld + k]; b.y = M[n * ld + k + 1];
  return b;
}
// D(16x16 tile at m0,n0) = A(.,K) x B(K,.) [+0], B optionally transposed
__device__ __forceinline__ void mm16(const float* A, int lda, const float* B, int ldb,
                                     float* D, int ldd, int m0, int n0, int K,
                                     bool btrans, int lane) {
  v8f acc = {};
  for (int k0 = 0; k0 < K; k0 += 4) {
    v2f av = ldA4(A, lda, m0, k0, lane);
    v2f bv = btrans ? ldBT4(B, ldb, k0, n0, lane) : ldB4(B, ldb, k0, n0, lane);
    acc = __builtin_amdgcn_wmma_f32_16x16x4_f32(false, av, false, bv,
                                                (short)0, acc, false, false);
  }
  int n  = n0 + (lane & 15);
  int mb = m0 + ((lane >> 4) << 3);
#pragma unroll
  for (int r = 0; r < 8; ++r) D[(mb + r) * ldd + n] = acc[r];
}

// ---------------- phase 2: Kalman filter, one wave32 per batch element ------
__global__ void __launch_bounds__(64)
kalman(const float* __restrict__ a, const float* __restrict__ Amats,
       const float* __restrict__ Cmats, const float* __restrict__ alpha,
       float* __restrict__ out) {
  __shared__ float sCm[2][16][32];
  __shared__ float sAn[2][32][32];
  __shared__ float sSg[2][32][32];
  __shared__ float sW1[2][32][32];
  __shared__ float sW2[2][32][32];
  __shared__ float sAug[2][16][32];
  __shared__ float sSC[2][32][16];
  __shared__ float sKg[2][32][16];
  __shared__ float sMu[2][32];
  __shared__ float sR[2][16];

  const int lane = threadIdx.x & 31;
  const int w    = threadIdx.x >> 5;
  const int b    = blockIdx.x * 2 + w;

  float (*Cm)[32]  = sCm[w];
  float (*An)[32]  = sAn[w];
  float (*Sg)[32]  = sSg[w];
  float (*W1)[32]  = sW1[w];
  float (*W2)[32]  = sW2[w];
  float (*Aug)[32] = sAug[w];
  float (*SC)[16]  = sSC[w];
  float (*Kg)[16]  = sKg[w];
  float* Mu = sMu[w];
  float* Rr = sR[w];

  for (int i = lane; i < 32 * 32; i += 32) {
    int r = i >> 5, c = i & 31;
    Sg[r][c] = (r == c) ? 1.0f : 0.0f;
  }
  Mu[lane] = 0.0f;
  __syncthreads();

  for (int t = 0; t < TT; ++t) {
    const float* al = alpha + ((size_t)b * TT + t) * KMIX;
    float w0 = al[0], w1v = al[1], w2v = al[2];
    int tn = (t + 1 < TT) ? (t + 1) : (TT - 1);
    const float* aln = alpha + ((size_t)b * TT + tn) * KMIX;
    float u0 = aln[0], u1 = aln[1], u2 = aln[2];
    for (int i = lane; i < 16 * 32; i += 32) {
      float v = w0  * Cmats[(size_t)t * 512 + i]
              + w1v * Cmats[(size_t)TT * 512 + (size_t)t * 512 + i]
              + w2v * Cmats[(size_t)2 * TT * 512 + (size_t)t * 512 + i];
      Cm[i >> 5][i & 31] = v;
    }
    for (int i = lane; i < 32 * 32; i += 32) {
      float v = u0 * Amats[(size_t)tn * 1024 + i]
              + u1 * Amats[(size_t)TT * 1024 + (size_t)tn * 1024 + i]
              + u2 * Amats[(size_t)2 * TT * 1024 + (size_t)tn * 1024 + i];
      An[i >> 5][i & 31] = v;
    }
    __syncthreads();

    // W1[0:16,:] = C @ Sigma
    mm16(&Cm[0][0], 32, &Sg[0][0], 32, &W1[0][0], 32, 0, 0,  32, false, lane);
    mm16(&Cm[0][0], 32, &Sg[0][0], 32, &W1[0][0], 32, 0, 16, 32, false, lane);
    __syncthreads();
    // Aug[:,0:16] = (C Sigma) C^T + R ; Aug[:,16:32] = I
    mm16(&W1[0][0], 32, &Cm[0][0], 32, &Aug[0][0], 32, 0, 0, 32, true, lane);
    __syncthreads();
    for (int i = lane; i < 16 * 16; i += 32) {
      int rI = i >> 4, cI = i & 15;
      Aug[rI][16 + cI] = (rI == cI) ? 1.0f : 0.0f;
      if (rI == cI) Aug[rI][cI] += 0.01f;
    }
    // residual r = a_t - C mu_pred
    if (lane < 16) {
      float s = a[((size_t)b * TT + t) * DAv + lane];
      for (int j = 0; j < DZv; ++j) s -= Cm[lane][j] * Mu[j];
      Rr[lane] = s;
    }
    __syncthreads();

    // Gauss-Jordan: Aug -> [I | S^-1]  (S is SPD: no pivoting needed)
    for (int p = 0; p < 16; ++p) {
      float inv = 1.0f / Aug[p][p];
      __syncthreads();
      Aug[p][lane] *= inv;
      __syncthreads();
      float fac[16];
#pragma unroll
      for (int rI = 0; rI < 16; ++rI) fac[rI] = Aug[rI][p];
      __syncthreads();
      float prow = Aug[p][lane];
#pragma unroll
      for (int rI = 0; rI < 16; ++rI)
        if (rI != p) Aug[rI][lane] -= fac[rI] * prow;
      __syncthreads();
    }

    // SC = Sigma @ C^T (32x16)
    mm16(&Sg[0][0], 32, &Cm[0][0], 32, &SC[0][0], 16, 0,  0, 32, true, lane);
    mm16(&Sg[0][0], 32, &Cm[0][0], 32, &SC[0][0], 16, 16, 0, 32, true, lane);
    __syncthreads();
    // Kg = SC @ S^-1 (32x16)
    mm16(&SC[0][0], 16, &Aug[0][16], 32, &Kg[0][0], 16, 0,  0, 16, false, lane);
    mm16(&SC[0][0], 16, &Aug[0][16], 32, &Kg[0][0], 16, 16, 0, 16, false, lane);
    __syncthreads();

    // mu_post = mu_pred + Kg r  (this is the output)
    float mupost = Mu[lane];
    for (int j = 0; j < 16; ++j) mupost += Kg[lane][j] * Rr[j];
    Mu[lane] = mupost;
    out[((size_t)b * TT + t) * DZv + lane] = mupost;
    __syncthreads();

    // W1 = Kg @ C (32x32) ; then W1 = I - W1
    mm16(&Kg[0][0], 16, &Cm[0][0], 32, &W1[0][0], 32, 0,  0,  16, false, lane);
    mm16(&Kg[0][0], 16, &Cm[0][0], 32, &W1[0][0], 32, 0,  16, 16, false, lane);
    mm16(&Kg[0][0], 16, &Cm[0][0], 32, &W1[0][0], 32, 16, 0,  16, false, lane);
    mm16(&Kg[0][0], 16, &Cm[0][0], 32, &W1[0][0], 32, 16, 16, 16, false, lane);
    __syncthreads();
    for (int i = lane; i < 32 * 32; i += 32) {
      int rI = i >> 5, cI = i & 31;
      W1[rI][cI] = ((rI == cI) ? 1.0f : 0.0f) - W1[rI][cI];
    }
    __syncthreads();
    // W2 = (I - Kg C) @ Sigma   (sigma posterior)
    mm16(&W1[0][0], 32, &Sg[0][0], 32, &W2[0][0], 32, 0,  0,  32, false, lane);
    mm16(&W1[0][0], 32, &Sg[0][0], 32, &W2[0][0], 32, 0,  16, 32, false, lane);
    mm16(&W1[0][0], 32, &Sg[0][0], 32, &W2[0][0], 32, 16, 0,  32, false, lane);
    mm16(&W1[0][0], 32, &Sg[0][0], 32, &W2[0][0], 32, 16, 16, 32, false, lane);
    __syncthreads();
    // W1 = A_next @ sigma_post
    mm16(&An[0][0], 32, &W2[0][0], 32, &W1[0][0], 32, 0,  0,  32, false, lane);
    mm16(&An[0][0], 32, &W2[0][0], 32, &W1[0][0], 32, 0,  16, 32, false, lane);
    mm16(&An[0][0], 32, &W2[0][0], 32, &W1[0][0], 32, 16, 0,  32, false, lane);
    mm16(&An[0][0], 32, &W2[0][0], 32, &W1[0][0], 32, 16, 16, 32, false, lane);
    __syncthreads();
    // mu_pred(next) = A_next @ mu_post  (read before Mu is overwritten)
    float mupred = 0.0f;
    for (int j = 0; j < DZv; ++j) mupred += An[lane][j] * Mu[j];
    // W2 = (A sigma_post) @ A^T
    mm16(&W1[0][0], 32, &An[0][0], 32, &W2[0][0], 32, 0,  0,  32, true, lane);
    mm16(&W1[0][0], 32, &An[0][0], 32, &W2[0][0], 32, 0,  16, 32, true, lane);
    mm16(&W1[0][0], 32, &An[0][0], 32, &W2[0][0], 32, 16, 0,  32, true, lane);
    mm16(&W1[0][0], 32, &An[0][0], 32, &W2[0][0], 32, 16, 16, 32, true, lane);
    __syncthreads();
    for (int i = lane; i < 32 * 32; i += 32) {
      int rI = i >> 5, cI = i & 31;
      Sg[rI][cI] = W2[rI][cI] + ((rI == cI) ? 0.01f : 0.0f);
    }
    Mu[lane] = mupred;
    __syncthreads();
  }
}

// ---------------------------------------------------------------------------
extern "C" void kernel_launch(void* const* d_in, const int* in_sizes, int n_in,
                              void* d_out, int out_size, void* d_ws, size_t ws_size,
                              hipStream_t stream) {
  const float* a    = (const float*)d_in[0];
  const float* A    = (const float*)d_in[1];
  const float* C    = (const float*)d_in[2];
  const float* a0   = (const float*)d_in[3];
  const float* Wih0 = (const float*)d_in[4];
  const float* Whh0 = (const float*)d_in[5];
  const float* bih0 = (const float*)d_in[6];
  const float* bhh0 = (const float*)d_in[7];
  const float* Wih1 = (const float*)d_in[8];
  const float* Whh1 = (const float*)d_in[9];
  const float* bih1 = (const float*)d_in[10];
  const float* bhh1 = (const float*)d_in[11];
  const float* Wlin = (const float*)d_in[12];
  const float* blin = (const float*)d_in[13];
  float* out = (float*)d_out;

  char* ws = (char*)d_ws;
  _Float16* packed = (_Float16*)ws;                      // 416*512 halves
  size_t off = (size_t)NFRAG * 512 * sizeof(_Float16);
  float* bias0 = (float*)(ws + off); off += G4 * sizeof(float);
  float* bias1 = (float*)(ws + off); off += G4 * sizeof(float);
  float* alpha = (float*)(ws + off);                     // BS*T*3 f32

  pack_weights<<<(NFRAG * 32 + 255) / 256, 256, 0, stream>>>(Wih0, Whh0, Wih1, Whh1, packed);
  pack_bias<<<2, 256, 0, stream>>>(bih0, bhh0, bih1, bhh1, bias0, bias1);
  lstm_alpha<<<BSv / 16, 1024, 0, stream>>>(a, a0, packed, bias0, bias1, Wlin, blin, alpha);
  kalman<<<BSv / 2, 64, 0, stream>>>(a, A, C, alpha, out);
}
